// Attention_75204877353064
// MI455X (gfx1250) — compile-verified, matrix operand order checked by reading, not run on hardware
//
#include <hip/hip_runtime.h>
#include <math.h>

// Problem constants (match reference)
#define B_  2
#define S_  2048
#define D_  4096
#define H_  32
#define HD_ 128

typedef __bf16 bf16x16 __attribute__((ext_vector_type(16)));
typedef float  f32x8   __attribute__((ext_vector_type(8)));

union Frag {
  uint4   u[2];   // two 16B loads
  bf16x16 v;      // 16 bf16 = 32B
  __bf16  e[16];
};

union Pack8 {     // 8 bf16 = one 16B store
  uint4  u;
  __bf16 e[8];
};

union Pack4 {     // 4 bf16 = one 8B store
  uint2  u;
  __bf16 e[4];
};

__device__ __forceinline__ f32x8 wmma_bf16(bf16x16 a, bf16x16 b, f32x8 c) {
  // (neg_a, A, neg_b, B, c_mod, C, reuse_a, reuse_b)
  return __builtin_amdgcn_wmma_f32_16x16x32_bf16(false, a, false, b, (short)0, c,
                                                 false, false);
}

// ---------------------------------------------------------------------------
// fp32 -> bf16 cast, 4-wide (n must be a multiple of 4 -- all our sizes are)
// ---------------------------------------------------------------------------
__global__ void cast_f32_bf16_v4(const float4* __restrict__ in,
                                 uint2* __restrict__ out, size_t n4) {
  size_t i = (size_t)blockIdx.x * blockDim.x + threadIdx.x;
  size_t stride = (size_t)gridDim.x * blockDim.x;
  for (; i < n4; i += stride) {
    float4 f = in[i];
    Pack4 p;
    p.e[0] = (__bf16)f.x;
    p.e[1] = (__bf16)f.y;
    p.e[2] = (__bf16)f.z;
    p.e[3] = (__bf16)f.w;
    out[i] = p.u;
  }
}

// ---------------------------------------------------------------------------
// C[M,N] = A[M,K] @ W[N,K]^T   (bf16 in, f32 accumulate, OutT out)
// Wave: 16x64 tile (A fragment reused across 4 B tiles); block = 4 waves.
// TR_OUT: store output transposed as (B, H, HD, S) -- used for V so the
// attention kernel's B-fragments (keys along the register axis) are
// contiguous 16-byte loads.
// ---------------------------------------------------------------------------
template <typename OutT, bool TR_OUT>
__global__ __launch_bounds__(128, 1) void gemm_xwT_wmma(
    const __bf16* __restrict__ A, const __bf16* __restrict__ W,
    OutT* __restrict__ C, int M, int N, int Kd) {
  const int lane = threadIdx.x & 31;
  const int wave = threadIdx.x >> 5;           // 0..3
  const int m0 = blockIdx.y * 64 + wave * 16;
  const int n0 = blockIdx.x * 64;
  const int lr = lane & 15;
  const int kh = (lane >> 4) << 3;             // 0 or 8 (K half-select)

  const __bf16* arow = A + (size_t)(m0 + lr) * Kd;
  const __bf16* wrow[4];
#pragma unroll
  for (int t = 0; t < 4; ++t) wrow[t] = W + (size_t)(n0 + t * 16 + lr) * Kd;

  f32x8 acc[4] = {};

  for (int k = 0; k < Kd; k += 32) {
    Frag af;
    af.u[0] = *(const uint4*)(arow + k + kh);
    af.u[1] = *(const uint4*)(arow + k + kh + 16);
    if (k + 512 < Kd) {
      __builtin_prefetch(arow + k + 512, 0, 1);
      __builtin_prefetch(wrow[0] + k + 512, 0, 1);
    }
#pragma unroll
    for (int t = 0; t < 4; ++t) {
      Frag bf;
      bf.u[0] = *(const uint4*)(wrow[t] + k + kh);
      bf.u[1] = *(const uint4*)(wrow[t] + k + kh + 16);
      acc[t] = wmma_bf16(af.v, bf.v, acc[t]);
    }
  }

  const int mb = m0 + ((lane >> 4) << 3);      // C layout rows r / r+8
  if constexpr (TR_OUT) {
    // (m,n) -> Vt[(b*H + h)*HD + hd][s], b=m/S, s=m%S, h=n/HD, hd=n%HD.
    // 8 consecutive r == 8 consecutive s: pack into one 16B store per tile.
    __bf16* ct = (__bf16*)C;
#pragma unroll
    for (int t = 0; t < 4; ++t) {
      const int n = n0 + t * 16 + lr;
      const size_t idx =
          ((size_t)((mb >> 11) * H_ + (n >> 7)) * HD_ + (n & (HD_ - 1))) * S_ +
          (mb & (S_ - 1));
      Pack8 p;
#pragma unroll
      for (int r = 0; r < 8; ++r) p.e[r] = (__bf16)acc[t][r];
      *(uint4*)(ct + idx) = p.u;
    }
  } else {
#pragma unroll
    for (int r = 0; r < 8; ++r) {
#pragma unroll
      for (int t = 0; t < 4; ++t) {
        C[(size_t)(mb + r) * N + n0 + t * 16 + lr] = (OutT)acc[t][r];
      }
    }
  }
}

// ---------------------------------------------------------------------------
// RoPE applied in place to q and k (bf16 storage, f32 math)
// ---------------------------------------------------------------------------
__global__ void rope_kernel(__bf16* __restrict__ q, __bf16* __restrict__ k,
                            const float* __restrict__ fcos,
                            const float* __restrict__ fsin) {
  const size_t total = (size_t)B_ * S_ * H_ * (HD_ / 2);
  size_t i = (size_t)blockIdx.x * blockDim.x + threadIdx.x;
  if (i >= total) return;
  const int p = (int)(i & (HD_ / 2 - 1));
  size_t r1 = i >> 6;                      // / (HD/2)
  const int h = (int)(r1 & (H_ - 1));
  size_t r2 = r1 >> 5;                     // / H
  const int s = (int)(r2 & (S_ - 1));
  const int b = (int)(r2 >> 11);           // / S
  const float c  = fcos[s * (HD_ / 2) + p];
  const float sn = fsin[s * (HD_ / 2) + p];
  const size_t base = (((size_t)b * S_ + s) * D_) + (size_t)h * HD_ + 2 * p;
  {
    float xr = (float)q[base], xi = (float)q[base + 1];
    q[base]     = (__bf16)(xr * c - xi * sn);
    q[base + 1] = (__bf16)(xr * sn + xi * c);
  }
  {
    float xr = (float)k[base], xi = (float)k[base + 1];
    k[base]     = (__bf16)(xr * c - xi * sn);
    k[base + 1] = (__bf16)(xr * sn + xi * c);
  }
}

// ---------------------------------------------------------------------------
// Flash attention: one wave per (b, h, 16-row q tile). Causal, online softmax.
// Q,K row-major (B,S,H,HD); V pre-transposed to (B,H,HD,S) so every WMMA
// operand is a pair of 16-byte global_load_b128.
// ---------------------------------------------------------------------------
__global__ __launch_bounds__(32, 1) void attn_kernel(
    const __bf16* __restrict__ Q, const __bf16* __restrict__ K,
    const __bf16* __restrict__ Vt, __bf16* __restrict__ O) {
  __shared__ float plds[16 * 33];

  const int lane = threadIdx.x;
  const int lr = lane & 15;
  const int kh = (lane >> 4) << 3;
  const int q0 = blockIdx.x * 16;
  const int h  = blockIdx.y;
  const int b  = blockIdx.z;
  const size_t bS = (size_t)b * S_;

  // Q fragments: 4 x (16x32) A-fragments covering HD=128
  Frag qf[4];
  {
    const __bf16* qrow = Q + (bS + q0 + lr) * D_ + (size_t)h * HD_;
#pragma unroll
    for (int c = 0; c < 4; ++c) {
      qf[c].u[0] = *(const uint4*)(qrow + c * 32 + kh);
      qf[c].u[1] = *(const uint4*)(qrow + c * 32 + kh + 16);
    }
  }

  f32x8 o[8] = {};
  float m_i[8], l_i[8];
#pragma unroll
  for (int r = 0; r < 8; ++r) { m_i[r] = -1e30f; l_i[r] = 0.0f; }

  const float scale = 0.088388347648318447f;   // 1/sqrt(128)
  const int mbq = (lane >> 4) << 3;            // 0 or 8: C-layout row offset
  const int ktiles = (q0 + 15) / 32 + 1;       // cover keys 0 .. q0+15

  const __bf16* vhead = Vt + ((size_t)b * H_ + h) * (size_t)HD_ * S_;

  for (int kt = 0; kt < ktiles; ++kt) {
    const int kbase = kt * 32;

    // --- scores: two 16x16 tiles, qf[c] consumed immediately per chunk ---
    f32x8 s0 = {}, s1 = {};
    {
      const __bf16* k0row = K + (bS + kbase + lr) * D_ + (size_t)h * HD_;
      const __bf16* k1row = k0row + (size_t)16 * D_;
#pragma unroll
      for (int c = 0; c < 4; ++c) {
        Frag kf0, kf1;
        kf0.u[0] = *(const uint4*)(k0row + c * 32 + kh);
        kf0.u[1] = *(const uint4*)(k0row + c * 32 + kh + 16);
        kf1.u[0] = *(const uint4*)(k1row + c * 32 + kh);
        kf1.u[1] = *(const uint4*)(k1row + c * 32 + kh + 16);
        s0 = wmma_bf16(qf[c].v, kf0.v, s0);
        s1 = wmma_bf16(qf[c].v, kf1.v, s1);
      }
    }

    // --- causal mask + online softmax stats ---
    float alpha_a[8], p0a[8], p1a[8];
#pragma unroll
    for (int r = 0; r < 8; ++r) {
      const int qr  = q0 + mbq + r;
      const int n0k = kbase + lr;
      float v0 = (n0k      <= qr) ? s0[r] * scale : -1e30f;
      float v1 = (n0k + 16 <= qr) ? s1[r] * scale : -1e30f;
      float mx = fmaxf(v0, v1);
#pragma unroll
      for (int off = 8; off >= 1; off >>= 1)   // reduce within 16-lane N group
        mx = fmaxf(mx, __shfl_xor(mx, off, 32));
      const float mnew  = fmaxf(m_i[r], mx);
      const float alpha = __expf(m_i[r] - mnew);
      const float e0 = __expf(v0 - mnew);
      const float e1 = __expf(v1 - mnew);
      float ps = e0 + e1;
#pragma unroll
      for (int off = 8; off >= 1; off >>= 1)
        ps += __shfl_xor(ps, off, 32);
      l_i[r] = l_i[r] * alpha + ps;
      m_i[r] = mnew;
      alpha_a[r] = alpha;
      p0a[r] = e0;
      p1a[r] = e1;
    }

    // --- rescale accumulators (separate tight loop, short live ranges) ---
#pragma unroll
    for (int t = 0; t < 8; ++t) {
#pragma unroll
      for (int r = 0; r < 8; ++r) o[t][r] *= alpha_a[r];
    }

    // --- transpose P from C layout to A layout through LDS ---
    __syncthreads();
#pragma unroll
    for (int r = 0; r < 8; ++r) {
      plds[(mbq + r) * 33 + lr]      = p0a[r];
      plds[(mbq + r) * 33 + 16 + lr] = p1a[r];
    }
    __syncthreads();
    Frag pf;
#pragma unroll
    for (int j = 0; j < 8; ++j) {
      pf.e[j]     = (__bf16)plds[lr * 33 + kh + j];
      pf.e[8 + j] = (__bf16)plds[lr * 33 + kh + 16 + j];
    }

    // --- PV: V^T rows are key-contiguous -> b128 fragment loads ---
#pragma unroll
    for (int t = 0; t < 8; ++t) {
      const __bf16* vc = vhead + (size_t)(t * 16 + lr) * S_ + kbase;
      Frag vf;
      vf.u[0] = *(const uint4*)(vc + kh);
      vf.u[1] = *(const uint4*)(vc + kh + 16);
      o[t] = wmma_bf16(pf.v, vf.v, o[t]);
    }
  }

  // --- normalize and store (bf16, (B,S,H,HD) contiguous == (B,S,D)) ---
  float rinv[8];
#pragma unroll
  for (int r = 0; r < 8; ++r) rinv[r] = 1.0f / l_i[r];
  __bf16* orow = O + (bS + q0 + mbq) * D_ + (size_t)h * HD_ + lr;
#pragma unroll
  for (int t = 0; t < 8; ++t) {
#pragma unroll
    for (int r = 0; r < 8; ++r) {
      orow[(size_t)r * D_ + t * 16] = (__bf16)(o[t][r] * rinv[r]);
    }
  }
}

// ---------------------------------------------------------------------------
extern "C" void kernel_launch(void* const* d_in, const int* in_sizes, int n_in,
                              void* d_out, int out_size, void* d_ws, size_t ws_size,
                              hipStream_t stream) {
  (void)in_sizes; (void)n_in; (void)out_size; (void)ws_size;

  const float* x  = (const float*)d_in[0];
  const float* wq = (const float*)d_in[1];
  const float* wk = (const float*)d_in[2];
  const float* wv = (const float*)d_in[3];
  const float* wo = (const float*)d_in[4];
  const float* fc = (const float*)d_in[5];
  const float* fs = (const float*)d_in[6];
  // d_in[7] (dense additive mask) unused: causal mask computed analytically.

  const size_t nX = (size_t)B_ * S_ * D_;   // 16,777,216 elements
  const size_t slab = nX * sizeof(__bf16);  // == D*D*2 as well

  char* ws = (char*)d_ws;
  __bf16* xbf = (__bf16*)(ws + 0 * slab);
  __bf16* wqb = (__bf16*)(ws + 1 * slab);
  __bf16* wkb = (__bf16*)(ws + 2 * slab);
  __bf16* wvb = (__bf16*)(ws + 3 * slab);
  __bf16* wob = (__bf16*)(ws + 4 * slab);
  __bf16* qb  = (__bf16*)(ws + 5 * slab);
  __bf16* kb  = (__bf16*)(ws + 6 * slab);
  __bf16* vtb = (__bf16*)(ws + 7 * slab);   // V stored transposed (B,H,HD,S)
  __bf16* ab  = (__bf16*)(ws + 8 * slab);

  // 1) precision casts (vectorized x4)
  cast_f32_bf16_v4<<<1024, 256, 0, stream>>>((const float4*)x,  (uint2*)xbf, nX / 4);
  cast_f32_bf16_v4<<<1024, 256, 0, stream>>>((const float4*)wq, (uint2*)wqb, nX / 4);
  cast_f32_bf16_v4<<<1024, 256, 0, stream>>>((const float4*)wk, (uint2*)wkb, nX / 4);
  cast_f32_bf16_v4<<<1024, 256, 0, stream>>>((const float4*)wv, (uint2*)wvb, nX / 4);
  cast_f32_bf16_v4<<<1024, 256, 0, stream>>>((const float4*)wo, (uint2*)wob, nX / 4);

  // 2) q/k/v projections: (B*S, D) @ (D, D)^T ; V stored transposed
  const dim3 gblock(128);
  const dim3 ggrid(D_ / 64, (B_ * S_) / 64);
  gemm_xwT_wmma<__bf16, false><<<ggrid, gblock, 0, stream>>>(xbf, wqb, qb,
                                                             B_ * S_, D_, D_);
  gemm_xwT_wmma<__bf16, false><<<ggrid, gblock, 0, stream>>>(xbf, wkb, kb,
                                                             B_ * S_, D_, D_);
  gemm_xwT_wmma<__bf16, true><<<ggrid, gblock, 0, stream>>>(xbf, wvb, vtb,
                                                            B_ * S_, D_, D_);

  // 3) RoPE on q, k
  {
    const size_t total = (size_t)B_ * S_ * H_ * (HD_ / 2);
    rope_kernel<<<(unsigned)((total + 255) / 256), 256, 0, stream>>>(qb, kb, fc, fs);
  }

  // 4) causal flash attention
  attn_kernel<<<dim3(S_ / 16, H_, B_), 32, 0, stream>>>(qb, kb, vtb, ab);

  // 5) output projection -> f32 d_out
  gemm_xwT_wmma<float, false><<<ggrid, gblock, 0, stream>>>(ab, wob, (float*)d_out,
                                                            B_ * S_, D_, D_);
}
